// MultiHeadAttention_53395033424058
// MI455X (gfx1250) — matrix operand order repaired and checked by reading, not compiled, not run on hardware
//
#include <hip/hip_runtime.h>
#include <hip/hip_bf16.h>

// ---------------------------------------------------------------------------
// MHA forward for MI455X (gfx1250, wave32, WMMA).
//   q = Q Wq^T + bq (scaled by log2(e)/8 -> base-2 softmax, f16, [B*H,S,64])
//   k = K Wk^T + bk (f16, [B*H, S, 64])
//   vT= (V Wv^T + bv)^T per head (f16, [B*H, 64, S])
//   flash attention with MULTIPLICATIVE tril mask (masked score = 0, still
//   participates in softmax, exactly like the reference). Causal split:
//     - tiles fully below diagonal: unmasked fast path (no cndmask)
//     - one boundary tile: masked path
//     - tiles fully above diagonal: all scores are exactly 0 => closed form
//       via per-head suffix sums of V (removes ~half the attention work)
//   out = Ctx Wo^T + bo (f32)
// All matmuls use v_wmma_f32_16x16x32_f16 (f32 accumulate). Cross-lane
// xor-16 moves use v_permlanex16 (VALU, no LDS pipe); softmax exponentials
// use raw v_exp_f32 via __builtin_amdgcn_exp2f (no libm range fix-up).
// ---------------------------------------------------------------------------

typedef __attribute__((ext_vector_type(16))) _Float16 v16h;
typedef __attribute__((ext_vector_type(8)))  _Float16 v8h;
typedef __attribute__((ext_vector_type(8)))  float    v8f;
typedef __attribute__((ext_vector_type(4)))  float    v4f;

union V16H { v16h v; v8h h[2]; unsigned u[8]; _Float16 e[16]; };

#define D_MODEL 1024
#define NHEAD   16
#define KD      64
#define BATCH   4
#define SEQ     1024
#define MROWS   (BATCH * SEQ)          // 4096
// q scale: 1/sqrt(64) * log2(e)  -> WMMA scores are already base-2 exponents
#define QSCALE  0.180336880111120429f

static __device__ __forceinline__ v8f wmma_f16(v16h a, v16h b, v8f c) {
  return __builtin_amdgcn_wmma_f32_16x16x32_f16(false, a, false, b, (short)0, c,
                                                false, false);
}

// raw v_exp_f32 (args are always <= 0 here; deep-negative flushes to 0, which
// is the correct softmax weight -- no libm range-reduction needed)
static __device__ __forceinline__ float fast_exp2(float x) {
#if __has_builtin(__builtin_amdgcn_exp2f)
  return __builtin_amdgcn_exp2f(x);
#else
  return exp2f(x);
#endif
}

// lane i <-> lane i^16 exchange (wave32): VALU permlane if available.
static __device__ __forceinline__ unsigned xor16_u32(unsigned x) {
#if __has_builtin(__builtin_amdgcn_permlanex16)
  return (unsigned)__builtin_amdgcn_permlanex16(
      (int)x, (int)x, 0x76543210, 0xfedcba98, false, false);
#else
  return (unsigned)__shfl_xor((int)x, 16, 32);
#endif
}
static __device__ __forceinline__ float xor16_f32(float x) {
  union { float f; unsigned u; } c;
  c.f = x;
  c.u = xor16_u32(c.u);
  return c.f;
}
// pack two floats into f16x2 (v_cvt_pk_f16_f32)
static __device__ __forceinline__ unsigned pk_f16(float a, float b) {
  union { _Float16 h[2]; unsigned u; } p;
  p.h[0] = (_Float16)a;
  p.h[1] = (_Float16)b;
  return p.u;
}

// ---------------- fp32 -> f16 weight conversion --------------------------
__global__ void cvt_f32_f16(const float* __restrict__ src,
                            _Float16* __restrict__ dst, int n) {
  int i = blockIdx.x * blockDim.x + threadIdx.x;
  if (i < n) dst[i] = (_Float16)src[i];
}

// ---------------- GEMM:  Y = X @ W^T + b ---------------------------------
// One wave computes a 16x32 tile of Y (A reused across 2 WMMAs per K-step).
// block = 4 waves -> 16x128 tile of Y.
// MODE 0: q  -> dstH[((b*16+h)*S+s)*64+d], scaled by QSCALE
// MODE 1: k  -> dstH[((b*16+h)*S+s)*64+d]
// MODE 2: v  -> dstH[((b*16+h)*64+d)*S+s]   (transposed for PV A-operand)
// MODE 3: Xh (f16 context) @ Wo^T + b -> dstF[m*1024+n]  (fp32 output)
template <int MODE>
__global__ __launch_bounds__(128)
void proj_gemm(const float* __restrict__ Xf, const _Float16* __restrict__ Xh,
               const _Float16* __restrict__ Wh, const float* __restrict__ bias,
               _Float16* __restrict__ dstH, float* __restrict__ dstF) {
  const int lane  = threadIdx.x & 31;
  const int wave  = threadIdx.x >> 5;
  const int l16   = lane & 15;
  const int lhi   = (lane >> 4) & 1;            // 0: lanes 0-15, 1: lanes 16-31
  const int n0    = blockIdx.x * 128 + wave * 32;
  const int m0    = blockIdx.y * 16;

  const int arow  = m0 + l16;                   // X row for A operand
  const int bcol0 = n0 + l16;                   // W rows (= Y columns)
  const int bcol1 = n0 + 16 + l16;

  v8f c0, c1;
  {
    const float bv0 = bias[bcol0];
    const float bv1 = bias[bcol1];
    #pragma unroll
    for (int i = 0; i < 8; ++i) { c0[i] = bv0; c1[i] = bv1; }
  }

  const float*    apf = (MODE < 3) ? Xf + (size_t)arow * D_MODEL : nullptr;
  const _Float16* aph = (MODE == 3) ? Xh + (size_t)arow * D_MODEL : nullptr;
  const _Float16* bp0 = Wh + (size_t)bcol0 * D_MODEL + lhi * 16;
  const _Float16* bp1 = Wh + (size_t)bcol1 * D_MODEL + lhi * 16;

  #pragma unroll 2
  for (int kk = 0; kk < D_MODEL; kk += 32) {
    V16H a, b0, b1;
    const int aoff = kk + lhi * 16;             // A: k = aoff+0..15
    if constexpr (MODE < 3) {
      v4f x0 = *(const v4f*)(apf + aoff);
      v4f x1 = *(const v4f*)(apf + aoff + 4);
      v4f x2 = *(const v4f*)(apf + aoff + 8);
      v4f x3 = *(const v4f*)(apf + aoff + 12);
      #pragma unroll
      for (int i = 0; i < 4; ++i) {
        a.e[i]      = (_Float16)x0[i];
        a.e[4 + i]  = (_Float16)x1[i];
        a.e[8 + i]  = (_Float16)x2[i];
        a.e[12 + i] = (_Float16)x3[i];
      }
    } else {
      a.h[0] = *(const v8h*)(aph + aoff);
      a.h[1] = *(const v8h*)(aph + aoff + 8);
    }
    // B[k][n] = W[n][k]: lane holds column n, 16 consecutive k (32B run)
    b0.v = *(const v16h*)(bp0 + kk);
    b1.v = *(const v16h*)(bp1 + kk);
    c0 = wmma_f16(a.v, b0.v, c0);
    c1 = wmma_f16(a.v, b1.v, c1);
  }

  // C layout: vgpr r -> row m0 + r + lhi*8 ; column = bcol
  const int mfirst = m0 + lhi * 8;
  #pragma unroll
  for (int t = 0; t < 2; ++t) {
    const v8f& c   = t ? c1 : c0;
    const int bcol = t ? bcol1 : bcol0;
    if constexpr (MODE == 3) {
      #pragma unroll
      for (int i = 0; i < 8; ++i)
        dstF[(size_t)(mfirst + i) * D_MODEL + bcol] = c[i];
    } else if constexpr (MODE == 2) {
      const int b_ = mfirst >> 10, s = mfirst & (SEQ - 1);
      const int h  = bcol >> 6,    d = bcol & (KD - 1);
      v8h out;
      #pragma unroll
      for (int i = 0; i < 8; ++i) out[i] = (_Float16)c[i];
      *(v8h*)(dstH + ((size_t)(b_ * NHEAD + h) * KD + d) * SEQ + s) = out;
    } else {
      const float scale = (MODE == 0) ? QSCALE : 1.0f;
      const int h = bcol >> 6, d = bcol & (KD - 1);
      #pragma unroll
      for (int i = 0; i < 8; ++i) {
        const int m  = mfirst + i;
        const int b_ = m >> 10, s = m & (SEQ - 1);
        dstH[((size_t)(b_ * NHEAD + h) * SEQ + s) * KD + d] =
            (_Float16)(c[i] * scale);
      }
    }
  }
}

// ---------------- per-head V suffix sums (tile granularity) --------------
// suffV[(bh*33 + t)*64 + d] = sum_{j >= 32t} vT[bh][d][j]   (f32)
__global__ __launch_bounds__(256)
void v_suffix(const _Float16* __restrict__ vT, float* __restrict__ suffV) {
  const int id = blockIdx.x * blockDim.x + threadIdx.x;   // 4096 threads
  const int bh = id >> 6, d = id & (KD - 1);
  const _Float16* vp = vT + ((size_t)bh * KD + d) * SEQ;
  float* sp = suffV + (size_t)bh * 33 * KD + d;
  float run = 0.0f;
  sp[32 * KD] = 0.0f;
  for (int t = 31; t >= 0; --t) {
    const _Float16* q = vp + t * 32;
    float s = 0.0f;
    #pragma unroll
    for (int j = 0; j < 32; ++j) s += (float)q[j];
    run += s;
    sp[t * KD] = run;
  }
}

// ---------------- one 32-key attention tile ------------------------------
// Scores from WMMA are base-2 exponents (QSCALE folded into q).
template <bool MASKED>
static __device__ __forceinline__ void attn_tile(
    int jbase, int thr, const _Float16* __restrict__ khead,
    const _Float16* __restrict__ vhead, v16h bq0, v16h bq1, int l16, int lhi,
    float& mrun, float& lrun, v8f acc[4]) {
  // ---- scores S^T: D0 holds j = jbase..+15, D1 holds j = jbase+16..+31
  v8f d0 = {}, d1 = {};
  {
    const _Float16* kb = khead + (size_t)(jbase + l16) * KD + lhi * 16;
    V16H a0, a1;
    a0.h[0] = *(const v8h*)(kb);        a0.h[1] = *(const v8h*)(kb + 8);
    a1.h[0] = *(const v8h*)(kb + 32);   a1.h[1] = *(const v8h*)(kb + 40);
    d0 = wmma_f16(a0.v, bq0, d0);
    d0 = wmma_f16(a1.v, bq1, d0);
    kb += 16 * KD;                      // rows j+16
    a0.h[0] = *(const v8h*)(kb);        a0.h[1] = *(const v8h*)(kb + 8);
    a1.h[0] = *(const v8h*)(kb + 32);   a1.h[1] = *(const v8h*)(kb + 40);
    d1 = wmma_f16(a0.v, bq0, d1);
    d1 = wmma_f16(a1.v, bq1, d1);
  }
  // ---- (mask +) online softmax, base 2; per-lane column = query iq
  float p0[8], p1[8];
  float lmax = -1e30f;
  #pragma unroll
  for (int r = 0; r < 8; ++r) {
    if constexpr (MASKED) {
      const int jl = r + lhi * 8;              // local j of d0 element
      p0[r] = (jl > thr) ? 0.0f : d0[r];       // masked score := 0 (NOT -inf)
      p1[r] = (jl + 16 > thr) ? 0.0f : d1[r];
    } else {
      p0[r] = d0[r];
      p1[r] = d1[r];
    }
    lmax = fmaxf(lmax, fmaxf(p0[r], p1[r]));
  }
  lmax = fmaxf(lmax, xor16_f32(lmax));
  const float mnew = fmaxf(mrun, lmax);
  const float corr = fast_exp2(mrun - mnew);
  float rs = 0.0f;
  #pragma unroll
  for (int r = 0; r < 8; ++r) {
    p0[r] = fast_exp2(p0[r] - mnew);
    p1[r] = fast_exp2(p1[r] - mnew);
    rs += p0[r] + p1[r];
  }
  rs += xor16_f32(rs);
  lrun = lrun * corr + rs;
  mrun = mnew;
  #pragma unroll
  for (int t = 0; t < 4; ++t)
    #pragma unroll
    for (int r = 0; r < 8; ++r) acc[t][r] *= corr;

  // ---- rebuild P^T as B operand. Pack f16 pairs FIRST, then one xor-16
  //      permlane per pair: 8 cross-lane moves instead of 16.
  //      lane<16 column needs j=0..15: own p0 pairs ++ partner p0 pairs.
  //      lane>=16 column needs j=16..31: partner p1 pairs ++ own p1 pairs.
  V16H bp;
  #pragma unroll
  for (int t = 0; t < 4; ++t) {
    const unsigned lo = pk_f16(p0[2 * t], p0[2 * t + 1]);   // j pair (own)
    const unsigned hi = pk_f16(p1[2 * t], p1[2 * t + 1]);
    const unsigned lo_x = xor16_u32(lo);
    const unsigned hi_x = xor16_u32(hi);
    bp.u[t]     = lhi ? hi_x : lo;     // seq pos 2t,2t+1
    bp.u[4 + t] = lhi ? hi : lo_x;     // seq pos 8+2t, 9+2t
  }
  // ---- O^T += V^T · P^T  (A = V^T rows d, k = j; contiguous in vT)
  const _Float16* vb = vhead + (size_t)l16 * SEQ + jbase + lhi * 16;
  #pragma unroll
  for (int dc = 0; dc < 4; ++dc) {
    const _Float16* vp = vb + (size_t)dc * 16 * SEQ;
    V16H av;
    av.h[0] = *(const v8h*)(vp);
    av.h[1] = *(const v8h*)(vp + 8);
    acc[dc] = wmma_f16(av.v, bp.v, acc[dc]);
  }
}

// ---------------- flash attention (transposed formulation) ---------------
// One wave owns 16 query rows of one (b,h). S^T = K·Q^T so the query index
// is the lane dimension: softmax row stats are per-lane scalars (one xor-16
// combine). Causal 3-phase split (see header comment).
__global__ __launch_bounds__(128)
void flash_attn(const _Float16* __restrict__ qh, const _Float16* __restrict__ kh,
                const _Float16* __restrict__ vT, const float* __restrict__ suffV,
                _Float16* __restrict__ Ctx) {
  const int lane = threadIdx.x & 31;
  const int l16  = lane & 15;
  const int lhi  = (lane >> 4) & 1;
  const int wid  = blockIdx.x * 4 + (threadIdx.x >> 5);
  const int bh   = wid >> 6;                 // 0..63  (b*16+h)
  const int i0   = (wid & 63) << 4;          // query tile start
  const int iq   = i0 + l16;                 // this lane's query index

  const _Float16* khead = kh + (size_t)bh * SEQ * KD;
  const _Float16* vhead = vT + (size_t)bh * KD * SEQ;

  // B operand (Q^T): column m = iq, k-dim = d (two chunks of 32)
  const _Float16* qbase = qh + ((size_t)bh * SEQ + iq) * KD;
  const v16h bq0 = *(const v16h*)(qbase + lhi * 16);
  const v16h bq1 = *(const v16h*)(qbase + 32 + lhi * 16);

  v8f acc[4];
  #pragma unroll
  for (int t = 0; t < 4; ++t) acc[t] = {};
  float mrun = -1e30f, lrun = 0.0f;

  const int tb = i0 >> 5;                    // the single boundary tile
  // phase 1: fully unmasked tiles (no compares, no cndmask)
  for (int kt = 0; kt < tb; ++kt)
    attn_tile<false>(kt * 32, 0, khead, vhead, bq0, bq1, l16, lhi,
                     mrun, lrun, acc);
  // phase 2: the one diagonal tile
  attn_tile<true>(tb * 32, iq - tb * 32, khead, vhead, bq0, bq1, l16, lhi,
                  mrun, lrun, acc);
  // phase 3: fully masked tail -- every score is exactly 0, weight 2^(-m).
  // contribution = 2^(-m) * suffix_sum(V); exact, no WMMA/exp per element.
  const int n3 = SEQ - (tb + 1) * 32;
  if (n3 > 0) {
    const float mnew = fmaxf(mrun, 0.0f);
    const float corr = fast_exp2(mrun - mnew);
    const float p    = fast_exp2(-mnew);
    lrun = lrun * corr + (float)n3 * p;
    mrun = mnew;
    const float* sp = suffV + ((size_t)bh * 33 + (tb + 1)) * KD + lhi * 8;
    #pragma unroll
    for (int dc = 0; dc < 4; ++dc) {
      const v8f sv = *(const v8f*)(sp + dc * 16);
      #pragma unroll
      for (int r = 0; r < 8; ++r) acc[dc][r] = acc[dc][r] * corr + p * sv[r];
    }
  }

  // ---- epilogue: divide by denom, store context [b, s, h*64+d] as f16
  const float inv = 1.0f / lrun;            // >= 1 term always present
  const int b_ = bh >> 4, h = bh & 15;
  const size_t row = ((size_t)b_ * SEQ + iq) * D_MODEL + h * KD;
  #pragma unroll
  for (int dc = 0; dc < 4; ++dc) {
    v8h out;
    #pragma unroll
    for (int r = 0; r < 8; ++r) out[r] = (_Float16)(acc[dc][r] * inv);
    *(v8h*)(Ctx + row + dc * 16 + lhi * 8) = out;
  }
}

// ---------------------------------------------------------------------------
extern "C" void kernel_launch(void* const* d_in, const int* in_sizes, int n_in,
                              void* d_out, int out_size, void* d_ws,
                              size_t ws_size, hipStream_t stream) {
  const float* V    = (const float*)d_in[0];
  const float* Q    = (const float*)d_in[1];
  const float* Kin  = (const float*)d_in[2];
  // d_in[3] = mask (tril) -- semantics reproduced analytically
  const float* Wq_w = (const float*)d_in[4];
  const float* Wq_b = (const float*)d_in[5];
  const float* Wk_w = (const float*)d_in[6];
  const float* Wk_b = (const float*)d_in[7];
  const float* Wv_w = (const float*)d_in[8];
  const float* Wv_b = (const float*)d_in[9];
  const float* Wo_w = (const float*)d_in[10];
  const float* Wo_b = (const float*)d_in[11];
  float* out = (float*)d_out;

  _Float16* ws = (_Float16*)d_ws;
  const size_t WELEM = (size_t)D_MODEL * D_MODEL;        // 1M
  const size_t HELEM = (size_t)BATCH * NHEAD * SEQ * KD; // 4M
  _Float16* WhQ = ws;
  _Float16* WhK = ws + WELEM;
  _Float16* WhV = ws + 2 * WELEM;
  _Float16* WhO = ws + 3 * WELEM;
  _Float16* qh  = ws + 4 * WELEM;            // [B*H, S, 64] (QSCALE folded)
  _Float16* kh  = qh + HELEM;                // [B*H, S, 64]
  _Float16* vT  = kh + HELEM;                // [B*H, 64, S]
  _Float16* Ctx = vT + HELEM;                // [B, S, 1024]
  float*   suffV = (float*)(Ctx + (size_t)MROWS * D_MODEL); // [64][33][64] f32
  // total: 40 MB + 0.54 MB of d_ws

  // 1) weights -> f16 (reused 256x each; converting once is the cheap path)
  {
    dim3 blk(256), grd((unsigned)((WELEM + 255) / 256));
    cvt_f32_f16<<<grd, blk, 0, stream>>>(Wq_w, WhQ, (int)WELEM);
    cvt_f32_f16<<<grd, blk, 0, stream>>>(Wk_w, WhK, (int)WELEM);
    cvt_f32_f16<<<grd, blk, 0, stream>>>(Wv_w, WhV, (int)WELEM);
    cvt_f32_f16<<<grd, blk, 0, stream>>>(Wo_w, WhO, (int)WELEM);
  }
  // 2) projections (fp32 activations converted in-register, WMMA f16 accum f32)
  {
    dim3 blk(128), grd(D_MODEL / 128, MROWS / 16);       // (8, 256)
    proj_gemm<0><<<grd, blk, 0, stream>>>(Q,   nullptr, WhQ, Wq_b, qh, nullptr);
    proj_gemm<1><<<grd, blk, 0, stream>>>(Kin, nullptr, WhK, Wk_b, kh, nullptr);
    proj_gemm<2><<<grd, blk, 0, stream>>>(V,   nullptr, WhV, Wv_b, vT, nullptr);
  }
  // 2b) per-head suffix sums of V for the fully-masked tail
  v_suffix<<<dim3(16), dim3(256), 0, stream>>>(vT, suffV);
  // 3) attention: 4096 waves (one per 16 query rows per (b,h)), 4 waves/block
  flash_attn<<<dim3(BATCH * NHEAD * (SEQ / 16) / 4), dim3(128), 0, stream>>>(
      qh, kh, vT, suffV, Ctx);
  // 4) output projection -> fp32
  {
    dim3 blk(128), grd(D_MODEL / 128, MROWS / 16);
    proj_gemm<3><<<grd, blk, 0, stream>>>(nullptr, Ctx, WhO, Wo_b, nullptr, out);
  }
}
